// PhysicsInformedCtxLiquidNetworkModel_65773129171574
// MI455X (gfx1250) — matrix operand order repaired and checked by reading, not compiled, not run on hardware
//
#include <hip/hip_runtime.h>
#include <hip/hip_bf16.h>

#define HIDDEN 512
#define NAPP   20
#define SEQ    1024
#define ROWS   16          // batch rows per workgroup
#define NWAVES 16          // 512 threads
#define HSTR   520         // padded LDS row stride (elements): 1040B = 65*16B, banks step by 4
#define DT_C   0.1f

typedef __attribute__((ext_vector_type(16))) __bf16 bf16x16;
typedef __attribute__((ext_vector_type(8)))  __bf16 bf16x8;
typedef __attribute__((ext_vector_type(8)))  float  f32x8;

__device__ __forceinline__ float tanh_fast(float x) {
  float e = __expf(-2.0f * fabsf(x));
  float r = (1.0f - e) / (1.0f + e);
  return copysignf(r, x);
}

// ---------------------------------------------------------------------------
// Prep: rec_w (f32, row-major [K=512][N=512]) -> bf16 in WMMA B-fragment order.
// Fragment element index: ((ntile*16 + ktile)*32 + lane)*16 + e
// holds rec_w[ ktile*32 + (lane<16 ? 0 : 16) + e ][ ntile*16 + lane%16 ]
// so the hot loop loads 16 contiguous bf16 (32B) per lane per fragment.
// ---------------------------------------------------------------------------
__global__ void prep_recw(const float* __restrict__ rec_w, __bf16* __restrict__ wb) {
  int t    = blockIdx.x * blockDim.x + threadIdx.x;   // 0 .. 512*512-1
  int e    = t & 15;
  int lane = (t >> 4) & 31;
  int kt   = (t >> 9) & 15;
  int nt   = t >> 13;
  int k = kt * 32 + ((lane < 16) ? 0 : 16) + e;
  int n = nt * 16 + (lane & 15);
  wb[t] = (__bf16)rec_w[k * HIDDEN + n];
}

// ---------------------------------------------------------------------------
// Main: 16 workgroups x 512 threads. WG owns 16 batch rows; wave w owns
// hidden columns [w*32, w*32+32) (two 16-wide WMMA N-tiles).
// ---------------------------------------------------------------------------
__global__ __launch_bounds__(512, 1) void liquid_main(
    const float* __restrict__ x,      const float* __restrict__ ctx,
    const float* __restrict__ ce_w1,  const float* __restrict__ ce_b1,
    const float* __restrict__ ce_w2,  const float* __restrict__ ce_b2,
    const float* __restrict__ in_w,   const float* __restrict__ in_b,
    const float* __restrict__ tau,
    const float* __restrict__ intra_g, const float* __restrict__ intra_b,
    const float* __restrict__ norm_g,  const float* __restrict__ norm_b,
    const float* __restrict__ head_w,  const float* __restrict__ head_b,
    const __bf16* __restrict__ wb,     float* __restrict__ out)
{
  __shared__ __bf16 sh_h[ROWS * HSTR];   // h (bf16), row-major, padded
  __shared__ float  sh_f[ROWS * HSTR];   // final normalized h (f32)
  __shared__ float  sh_ps[NWAVES * 16];  // LN partial sums
  __shared__ float  sh_pq[NWAVES * 16];  // LN partial sum-of-squares
  __shared__ float  sh_x[ROWS * 2];      // x[:, t, :] slice for this WG
  __shared__ float  sh_rv[ROWS * 32];    // relu hidden of context encoder

  const int tid  = threadIdx.x;
  const int lane = tid & 31;
  const int w    = tid >> 5;
  const int rowBase = blockIdx.x * ROWS;
  const int nloc  = lane & 15;
  const int mofs  = (lane < 16) ? 0 : 8;   // C/D layout: m = j + mofs
  const int akofs = (lane < 16) ? 0 : 16;  // A layout:  K offset 0 or 16
  const int am    = nloc;                  // A layout:  M = lane%16
  const int n0 = w * 32 + nloc;
  const int n1 = n0 + 16;

  // per-lane hidden-column constants (invariant over t)
  const float w00 = in_w[n0], w01 = in_w[HIDDEN + n0], ib0 = in_b[n0];
  const float w10 = in_w[n1], w11 = in_w[HIDDEN + n1], ib1 = in_b[n1];
  const float ig0 = intra_g[n0], ibt0 = intra_b[n0];
  const float ig1 = intra_g[n1], ibt1 = intra_b[n1];
  const float ka0 = 1.0f - DT_C / log1pf(__expf(tau[n0]));  // 1 - DT/softplus
  const float ka1 = 1.0f - DT_C / log1pf(__expf(tau[n1]));

  // ---- context encoder stage 1: rv[m][j] = relu(ctx @ ce_w1 + ce_b1)
  {
    int m = tid >> 5, j = tid & 31;   // all 512 threads
    float s = ce_b1[j];
    #pragma unroll
    for (int i = 0; i < 6; ++i) s += ctx[(rowBase + m) * 6 + i] * ce_w1[i * 32 + j];
    sh_rv[m * 32 + j] = fmaxf(s, 0.0f);
  }
  __syncthreads();

  // ---- h0 = tanh(rv @ ce_w2 + ce_b2); keep in registers (C/D layout), mirror bf16 to LDS
  float h0v[8], h1v[8];
  #pragma unroll
  for (int j = 0; j < 8; ++j) {
    int m = j + mofs;
    float a0 = ce_b2[n0], a1 = ce_b2[n1];
    for (int q = 0; q < 32; ++q) {
      float r = sh_rv[m * 32 + q];
      a0 += r * ce_w2[q * HIDDEN + n0];
      a1 += r * ce_w2[q * HIDDEN + n1];
    }
    h0v[j] = tanh_fast(a0);
    h1v[j] = tanh_fast(a1);
    sh_h[m * HSTR + n0] = (__bf16)h0v[j];
    sh_h[m * HSTR + n1] = (__bf16)h1v[j];
  }
  __syncthreads();

  // Per-lane byte offset of this lane's (ntile0, kt=0) B fragment.
  // ntile0 = 2*w; fragment (nt,kt) lives at ((nt*16 + kt)*32 + lane)*32 bytes.
  int boff = ((2 * w) * 512 + lane) * 32;

  // ============================ 1024-step scan ============================
  for (int t = 0; t < SEQ; ++t) {
    if (tid < 32) {   // stage x[rows, t, 0..1]
      int m = tid >> 1;
      sh_x[tid] = x[((size_t)(rowBase + m) * SEQ + t) * 2 + (tid & 1)];
    }

    // Opaque no-op: "redefines" boff every iteration so the loop-invariant
    // B-fragment loads cannot be hoisted (which previously forced a 256-VGPR
    // spill to scratch). Keeps the weights streaming from the shared,
    // L2-resident swizzled buffer each step.
    asm volatile("" : "+v"(boff));
    const char* bp = (const char*)wb + boff;

    // rec = h @ rec_w  (bf16 WMMA, f32 accum): 16 K-tiles x 2 N-tiles
    f32x8 acc0 = {0.f,0.f,0.f,0.f,0.f,0.f,0.f,0.f};
    f32x8 acc1 = {0.f,0.f,0.f,0.f,0.f,0.f,0.f,0.f};
    #pragma unroll
    for (int kt = 0; kt < 16; ++kt) {
      bf16x8 alo = *(const bf16x8*)(&sh_h[am * HSTR + kt * 32 + akofs]);
      bf16x8 ahi = *(const bf16x8*)(&sh_h[am * HSTR + kt * 32 + akofs + 8]);
      bf16x16 a;
      #pragma unroll
      for (int i = 0; i < 8; ++i) { a[i] = alo[i]; a[i + 8] = ahi[i]; }
      bf16x16 b0 = *(const bf16x16*)(bp + kt * 1024);            // ntile0, kt
      bf16x16 b1 = *(const bf16x16*)(bp + 16384 + kt * 1024);    // ntile1, kt
      acc0 = __builtin_amdgcn_wmma_f32_16x16x32_bf16(false, a, false, b0, (short)0, acc0, false, false);
      acc1 = __builtin_amdgcn_wmma_f32_16x16x32_bf16(false, a, false, b1, (short)0, acc1, false, false);
    }
    __syncthreads();   // h reads done (safe to overwrite later); sh_x visible

    // u = xp + rec ; per-row LN partials
    float s[8], q[8];
    #pragma unroll
    for (int j = 0; j < 8; ++j) {
      int m = j + mofs;
      float xa = sh_x[2 * m], xb = sh_x[2 * m + 1];
      float u0 = acc0[j] + xa * w00 + xb * w01 + ib0;
      float u1 = acc1[j] + xa * w10 + xb * w11 + ib1;
      acc0[j] = u0; acc1[j] = u1;
      s[j] = u0 + u1;
      q[j] = u0 * u0 + u1 * u1;
    }
    #pragma unroll
    for (int j = 0; j < 8; ++j) {
      #pragma unroll
      for (int mk = 1; mk < 16; mk <<= 1) {
        s[j] += __shfl_xor(s[j], mk, 32);
        q[j] += __shfl_xor(q[j], mk, 32);
      }
    }
    if (nloc == 0) {   // lanes 0 and 16 publish this wave's 32-col partials
      #pragma unroll
      for (int j = 0; j < 8; ++j) {
        sh_ps[w * 16 + mofs + j] = s[j];
        sh_pq[w * 16 + mofs + j] = q[j];
      }
    }
    __syncthreads();

    float mean_r = 0.f, rstd_r = 0.f;
    if (lane < 16) {   // lane r finalizes row r stats
      float ts = 0.f, tq = 0.f;
      #pragma unroll
      for (int ww = 0; ww < NWAVES; ++ww) {
        ts += sh_ps[ww * 16 + lane];
        tq += sh_pq[ww * 16 + lane];
      }
      mean_r = ts * (1.0f / HIDDEN);
      float var = tq * (1.0f / HIDDEN) - mean_r * mean_r;
      rstd_r = rsqrtf(var + 1e-5f);
    }

    // f = tanh(LN(u)); h = clip(h*ka + DT*f); mirror bf16 h to LDS
    #pragma unroll
    for (int j = 0; j < 8; ++j) {
      int m = j + mofs;
      float mu = __shfl(mean_r, m, 32);
      float rs = __shfl(rstd_r, m, 32);
      float f0 = tanh_fast((acc0[j] - mu) * rs * ig0 + ibt0);
      float f1 = tanh_fast((acc1[j] - mu) * rs * ig1 + ibt1);
      float hn0 = fminf(fmaxf(h0v[j] * ka0 + f0 * DT_C, -10.f), 10.f);
      float hn1 = fminf(fmaxf(h1v[j] * ka1 + f1 * DT_C, -10.f), 10.f);
      h0v[j] = hn0; h1v[j] = hn1;
      sh_h[m * HSTR + n0] = (__bf16)hn0;
      sh_h[m * HSTR + n1] = (__bf16)hn1;
    }
    __syncthreads();   // h ready for next step's GEMM; partials free
  }

  // ======================= final LN + heads epilogue ======================
  const float ng0 = norm_g[n0], nb0 = norm_b[n0];
  const float ng1 = norm_g[n1], nb1 = norm_b[n1];
  {
    float s[8], q[8];
    #pragma unroll
    for (int j = 0; j < 8; ++j) {
      s[j] = h0v[j] + h1v[j];
      q[j] = h0v[j] * h0v[j] + h1v[j] * h1v[j];
      #pragma unroll
      for (int mk = 1; mk < 16; mk <<= 1) {
        s[j] += __shfl_xor(s[j], mk, 32);
        q[j] += __shfl_xor(q[j], mk, 32);
      }
    }
    if (nloc == 0) {
      #pragma unroll
      for (int j = 0; j < 8; ++j) {
        sh_ps[w * 16 + mofs + j] = s[j];
        sh_pq[w * 16 + mofs + j] = q[j];
      }
    }
    __syncthreads();
    float mean_r = 0.f, rstd_r = 0.f;
    if (lane < 16) {
      float ts = 0.f, tq = 0.f;
      #pragma unroll
      for (int ww = 0; ww < NWAVES; ++ww) {
        ts += sh_ps[ww * 16 + lane];
        tq += sh_pq[ww * 16 + lane];
      }
      mean_r = ts * (1.0f / HIDDEN);
      float var = tq * (1.0f / HIDDEN) - mean_r * mean_r;
      rstd_r = rsqrtf(var + 1e-5f);
    }
    #pragma unroll
    for (int j = 0; j < 8; ++j) {
      int m = j + mofs;
      float mu = __shfl(mean_r, m, 32);
      float rs = __shfl(rstd_r, m, 32);
      sh_f[m * HSTR + n0] = (h0v[j] - mu) * rs * ng0 + nb0;
      sh_f[m * HSTR + n1] = (h1v[j] - mu) * rs * ng1 + nb1;
    }
  }
  __syncthreads();

  if (tid < ROWS * NAPP) {   // 320 threads: out[m][p] = hn[m,:] . head_w[:,p]
    int m = tid / NAPP, p = tid % NAPP;
    float acc = head_b[p];
    for (int n = 0; n < HIDDEN; ++n)
      acc += sh_f[m * HSTR + n] * head_w[n * NAPP + p];
    out[(rowBase + m) * NAPP + p] = acc;
  }
}

extern "C" void kernel_launch(void* const* d_in, const int* in_sizes, int n_in,
                              void* d_out, int out_size, void* d_ws, size_t ws_size,
                              hipStream_t stream) {
  (void)in_sizes; (void)n_in; (void)out_size; (void)ws_size;
  const float* x       = (const float*)d_in[0];
  const float* ctx     = (const float*)d_in[1];
  const float* ce_w1   = (const float*)d_in[2];
  const float* ce_b1   = (const float*)d_in[3];
  const float* ce_w2   = (const float*)d_in[4];
  const float* ce_b2   = (const float*)d_in[5];
  const float* in_w    = (const float*)d_in[6];
  const float* in_b    = (const float*)d_in[7];
  const float* rec_w   = (const float*)d_in[8];
  const float* tau     = (const float*)d_in[9];
  const float* intra_g = (const float*)d_in[10];
  const float* intra_b = (const float*)d_in[11];
  const float* norm_g  = (const float*)d_in[12];
  const float* norm_b  = (const float*)d_in[13];
  const float* head_w  = (const float*)d_in[14];
  const float* head_b  = (const float*)d_in[15];

  __bf16* wb = (__bf16*)d_ws;   // 512 KB of B-fragment-swizzled bf16 rec_w

  prep_recw<<<512, 512, 0, stream>>>(rec_w, wb);
  liquid_main<<<16, 512, 0, stream>>>(x, ctx, ce_w1, ce_b1, ce_w2, ce_b2,
                                      in_w, in_b, tau, intra_g, intra_b,
                                      norm_g, norm_b, head_w, head_b,
                                      wb, (float*)d_out);
}
